// quantizer_kotha_42090679501587
// MI455X (gfx1250) — matrix operand order repaired and checked by reading, not compiled, not run on hardware
//
#include <hip/hip_runtime.h>
#include <hip/hip_bf16.h>
#include <math.h>

typedef __attribute__((ext_vector_type(16))) _Float16 v16h;
typedef __attribute__((ext_vector_type(8)))  float    v8f;

#define V_DIM   256   // vector length per code
#define K_CODES 512   // number of codebook entries

// ---------------------------------------------------------------------------
// Prep: f32 codebook -> f16 copy (WMMA B operand source), bias[k] = -0.5*||e_k||^2
// (so WMMA-accumulated score = x.e - 0.5||e||^2; argmax score == argmin dist).
// Also zeroes the histogram (runs before the main kernel on the same stream).
// ---------------------------------------------------------------------------
__global__ __launch_bounds__(256)
void vq_prep_kernel(const float* __restrict__ emb,
                    _Float16* __restrict__ embh,
                    float* __restrict__ ebias,
                    int* __restrict__ hist) {
  const int k = blockIdx.x;
  const int v = threadIdx.x;           // 0..255
  float e = emb[(size_t)k * V_DIM + v];
  embh[(size_t)k * V_DIM + v] = (_Float16)e;

  __shared__ float red[256];
  red[v] = e * e;
  __syncthreads();
  #pragma unroll
  for (int off = 128; off > 0; off >>= 1) {
    if (v < off) red[v] += red[v + off];
    __syncthreads();
  }
  if (v == 0) {
    ebias[k] = -0.5f * red[0];
    hist[k]  = 0;
  }
}

// ---------------------------------------------------------------------------
// Main fused kernel: per wave, 16 rows of x.
//   - build f16 A-fragments for x (ISA 16-bit 16x32 A layout), reused 32x
//   - 32 k-tiles x 8 WMMA (V_WMMA_F32_16X16X32_F16), C preloaded with bias
//   - per-lane argmax + cross-lane (16-lane half) argmax reduction
//   - gather winning code rows, write straight-through out0 = (e - x) + x,
//     squared-error sums out1/out2, histogram atomics
// ---------------------------------------------------------------------------
__global__ __launch_bounds__(256)
void vq_main_kernel(const float* __restrict__ x,
                    const float* __restrict__ emb,
                    const _Float16* __restrict__ embh,
                    const float* __restrict__ ebias,
                    float* __restrict__ out0,
                    float* __restrict__ out1,
                    float* __restrict__ out2,
                    int* __restrict__ hist) {
  const int lane = threadIdx.x & 31;
  const int wid  = threadIdx.x >> 5;
  const int m    = lane & 15;          // row within tile (A) / column (B,C)
  const int half = lane >> 4;          // K-half selector
  const int rowBase = blockIdx.x * 128 + wid * 16;

  union V16 { v16h v; _Float16 h[16]; unsigned u[8]; };

  // ---- A fragments: rows rowBase..rowBase+15 of x, f32 -> f16.
  // ISA 16-bit A 16x32 layout: lane m half 0 holds K {0..7,16..23},
  // half 1 holds K {8..15,24..31}; reg j packs K pair
  //   kb = 2*j + (j>=4 ? 8 : 0) + 8*half.
  V16 a[8];
  const float* xrow = x + (size_t)(rowBase + m) * V_DIM;
  #pragma unroll
  for (int vc = 0; vc < 8; ++vc) {
    #pragma unroll
    for (int j = 0; j < 8; ++j) {
      const int kb = vc * 32 + 2 * j + ((j >= 4) ? 8 : 0) + 8 * half;
      float2 p = *(const float2*)(xrow + kb);
      a[vc].h[2 * j]     = (_Float16)p.x;
      a[vc].h[2 * j + 1] = (_Float16)p.y;
    }
  }

  float best[8];
  int   bestk[8];
  #pragma unroll
  for (int r = 0; r < 8; ++r) { best[r] = -3.0e38f; bestk[r] = 0; }

  // ---- score all 512 codes in 32 tiles of 16 columns
  for (int kt = 0; kt < 32; ++kt) {
    const int col = kt * 16 + m;
    const float bias = ebias[col];
    v8f c;
    #pragma unroll
    for (int r = 0; r < 8; ++r) c[r] = bias;   // C := -0.5*||e_col||^2

    // B 32x16 f16 layout: lanes = columns (lane&15), reg j holds rows
    // {kb, kb+1}, kb = 2j + (j>=4 ? 8 : 0) + 8*half; pairs contiguous in embh -> one dword each.
    const _Float16* bcol = embh + (size_t)col * V_DIM;
    #pragma unroll
    for (int vc = 0; vc < 8; ++vc) {
      V16 b;
      #pragma unroll
      for (int j = 0; j < 8; ++j)
        b.u[j] = *(const unsigned*)(bcol + vc * 32 + 2 * j + ((j >= 4) ? 8 : 0) + 8 * half);
      c = __builtin_amdgcn_wmma_f32_16x16x32_f16(
              /*neg_a=*/false, a[vc].v, /*neg_b=*/false, b.v,
              /*c_mod=*/(short)0, c, /*reuse_a=*/false, /*reuse_b=*/false);
    }

    // strict '>' keeps the earliest (smallest) k on ties, matching argmin
    #pragma unroll
    for (int r = 0; r < 8; ++r)
      if (c[r] > best[r]) { best[r] = c[r]; bestk[r] = col; }
  }

  // ---- cross-lane argmax within each 16-lane half (xor 1,2,4,8 stays in half)
  #pragma unroll
  for (int r = 0; r < 8; ++r) {
    #pragma unroll
    for (int msk = 1; msk <= 8; msk <<= 1) {
      float ob = __shfl_xor(best[r],  msk, 32);
      int   ok = __shfl_xor(bestk[r], msk, 32);
      if (ob > best[r] || (ob == best[r] && ok < bestk[r])) {
        best[r] = ob; bestk[r] = ok;
      }
    }
  }
  // rows 0..7 -> lanes 0..15 reg r; rows 8..15 -> lanes 16..31 reg r-8

  // ---- outputs: 32 lanes cooperate per row (8 floats each)
  #pragma unroll
  for (int rr = 0; rr < 16; ++rr) {
    const int k   = __shfl(bestk[rr & 7], (rr < 8) ? 0 : 16, 32);
    const int row = rowBase + rr;
    const float* erow = emb + (size_t)k * V_DIM;
    const float* xr   = x   + (size_t)row * V_DIM;
    float* orow = out0 + (size_t)row * V_DIM;

    float acc = 0.0f;
    #pragma unroll
    for (int t = 0; t < 2; ++t) {
      const int idx = lane * 8 + t * 4;
      float4 e4 = *(const float4*)(erow + idx);
      float4 x4 = *(const float4*)(xr + idx);
      // straight-through: out0 = stop_grad(e - x) + x, evaluated in f32
      float4 o4;
      o4.x = (e4.x - x4.x) + x4.x;
      o4.y = (e4.y - x4.y) + x4.y;
      o4.z = (e4.z - x4.z) + x4.z;
      o4.w = (e4.w - x4.w) + x4.w;
      *(float4*)(orow + idx) = o4;
      float d0 = x4.x - e4.x, d1 = x4.y - e4.y;
      float d2 = x4.z - e4.z, d3 = x4.w - e4.w;
      acc += d0 * d0 + d1 * d1 + d2 * d2 + d3 * d3;
    }
    #pragma unroll
    for (int msk = 16; msk >= 1; msk >>= 1)
      acc += __shfl_xor(acc, msk, 32);
    if (lane == 0) {
      out1[row] = acc;       // sum((x - output)^2)
      out2[row] = acc;       // + sum((x - x0)^2) == 0 since x == x0
      atomicAdd(&hist[k], 1);
    }
  }
}

// ---------------------------------------------------------------------------
// Entropy over code usage: -sum_{hist>0} p*log(p), p = hist/N
// ---------------------------------------------------------------------------
__global__ __launch_bounds__(512)
void vq_entropy_kernel(const int* __restrict__ hist, float inv_n,
                       float* __restrict__ out_entropy) {
  __shared__ float red[512];
  const int k = threadIdx.x;
  const int h = hist[k];
  float val = 0.0f;
  if (h > 0) {
    float p = (float)h * inv_n;
    val = -p * logf(p);
  }
  red[k] = val;
  __syncthreads();
  #pragma unroll
  for (int off = 256; off > 0; off >>= 1) {
    if (k < off) red[k] += red[k + off];
    __syncthreads();
  }
  if (k == 0) *out_entropy = red[0];
}

extern "C" void kernel_launch(void* const* d_in, const int* in_sizes, int n_in,
                              void* d_out, int out_size, void* d_ws, size_t ws_size,
                              hipStream_t stream) {
  const float* x   = (const float*)d_in[0];   // (16,2048,1,256) f32
  const float* emb = (const float*)d_in[1];   // (1,512,256)     f32

  const int N = in_sizes[0] / V_DIM;          // 32768 rows
  const int K = in_sizes[1] / V_DIM;          // 512 codes

  // workspace: f16 codebook | bias[K] | hist[K]
  _Float16* embh  = (_Float16*)d_ws;
  float*    ebias = (float*)((char*)d_ws + (size_t)K * V_DIM * sizeof(_Float16));
  int*      hist  = (int*)(ebias + K);

  // outputs concatenated: out0 (N*V) | out1 (N) | out2 (N) | entropy (1)
  float* out0 = (float*)d_out;
  float* out1 = out0 + (size_t)N * V_DIM;
  float* out2 = out1 + N;
  float* oent = out2 + N;

  vq_prep_kernel<<<K, 256, 0, stream>>>(emb, embh, ebias, hist);
  vq_main_kernel<<<N / 128, 256, 0, stream>>>(x, emb, embh, ebias,
                                              out0, out1, out2, hist);
  vq_entropy_kernel<<<1, 512, 0, stream>>>(hist, 1.0f / (float)N, oent);
}
